// WordEncoder_4741643894793
// MI455X (gfx1250) — matrix operand order repaired and checked by reading, not compiled, not run on hardware
//
#include <hip/hip_runtime.h>

// ---------------------------------------------------------------------------
// Problem constants (from reference)
// ---------------------------------------------------------------------------
static constexpr int BB   = 32;
static constexpr int SS   = 512;
static constexpr int HH   = 768;
static constexpr int WW   = 128;
static constexpr int GG   = 4;
static constexpr int NH   = 4;
static constexpr int HD   = 192;
static constexpr int FF   = 2048;
static constexpr int NL   = 2;
static constexpr int NLAB = 6;
static constexpr int LW   = WW + 1;       // 129
static constexpr int MR   = BB * LW;      // 4128 rows of x

typedef __attribute__((ext_vector_type(16))) __bf16 v16bf;
typedef __attribute__((ext_vector_type(8)))  float  v8f;
typedef unsigned short u16;
typedef unsigned int   u32;
typedef unsigned char  u8;

__device__ __forceinline__ u16 f2bf(float f) {
    u32 x = __builtin_bit_cast(u32, f);
    x += 0x7FFFu + ((x >> 16) & 1u);      // round-to-nearest-even
    return (u16)(x >> 16);
}

// CDNA5 async global->LDS copy, 16 bytes per lane (tracked by ASYNCcnt).
// VGLOBAL encoding: VDST = LDS byte address VGPR, VADDR = 64-bit global addr.
__device__ __forceinline__ void async_copy_b128(const void* gsrc, void* ldst) {
    const unsigned lds  = (unsigned)(unsigned long long)ldst;   // low 32b = LDS offset
    const unsigned long long ga = (unsigned long long)gsrc;
    asm volatile("global_load_async_to_lds_b128 %0, %1, off"
                 :: "v"(lds), "v"(ga) : "memory");
}
__device__ __forceinline__ void wait_asynccnt0() {
    asm volatile("s_wait_asynccnt 0x0" ::: "memory");
}

// ---------------------------------------------------------------------------
// Generic batched GEMM:  C = act(alpha * A @ B^T + bias + residual)
//   A  : (M,K) bf16, row stride lda, batch offset zo*sAo + zi*sAi
//   Bm : "weight-like" operand, element (n,k) at n*ldbn + k*ldbk (bf16)
//   C  : (M,N) fp32 (outBf16==0) or bf16 (outBf16==1), row stride ldc
// 64x64 block tile, 4 waves (wave32), each wave 2x2 of 16x16 WMMA tiles,
// K-step 32, double-buffered LDS staged via global_load_async_to_lds_b128,
// compute with v_wmma_f32_16x16x32_bf16.
//
// OOB policy: A rows are clamped to M-1 (they only feed discarded output
// rows), B rows clamped to N-1 (feed discarded output cols); only the K tail
// needs true zero-fill, handled by the (wave-uniform) slow path. This keeps
// the async staging branch-free and EXEC fully populated.
// ---------------------------------------------------------------------------
__global__ __launch_bounds__(128)
void gemm_bf16_wmma(const u16* __restrict__ A,  long long lda,  long long sAo, long long sAi,
                    const u16* __restrict__ Bm, long long ldbn, long long ldbk,
                    long long sBo, long long sBi,
                    const float* __restrict__ bias,
                    const float* __restrict__ resid, long long ldr,
                    void* __restrict__ Cout, long long ldc, long long sCo, long long sCi,
                    int M, int N, int K, float alpha, int act, int outBf16, int batchInner)
{
    const int tid   = threadIdx.x;
    const int lane  = tid & 31;
    const int wv    = tid >> 5;
    const int waveM = wv >> 1;
    const int waveN = wv & 1;
    const int half  = lane >> 4;    // 0: lanes 0-15, 1: lanes 16-31
    const int l15   = lane & 15;

    const int zo = blockIdx.z / batchInner;
    const int zi = blockIdx.z % batchInner;
    A  += (long long)zo * sAo + (long long)zi * sAi;
    Bm += (long long)zo * sBo + (long long)zi * sBi;
    const long long cOff = (long long)zo * sCo + (long long)zi * sCi;
    float* Cf = (float*)Cout + cOff;
    u16*   Cb = (u16*)  Cout + cOff;

    const int tileM = blockIdx.y * 64;
    const int tileN = blockIdx.x * 64;

    // double-buffered tiles; rows are 80B (multiple of 16B) so b128 LDS
    // destinations and ds_load_b128 fragment reads stay 16B aligned.
    __shared__ __align__(16) u16 As[2][64][40];
    __shared__ __align__(16) u16 Bs[2][64][40];

    const v8f vzero = {0.f,0.f,0.f,0.f,0.f,0.f,0.f,0.f};
    v8f acc[4];
    acc[0] = vzero; acc[1] = vzero; acc[2] = vzero; acc[3] = vzero;

    const int sr = tid >> 1;          // staging row 0..63
    const int sc = (tid & 1) * 16;    // staging half-row: 0 or 16

    const bool fastA = ((lda & 7) == 0);
    const bool fastB = (ldbk == 1) && ((ldbn & 7) == 0);

    // row-clamped source rows (always in-bounds; OOB rows feed discarded C)
    const int gmA = tileM + sr;
    const int gnB = tileN + sr;
    const long long rowA = (gmA < M) ? gmA : (M - 1);
    const long long rowB = (gnB < N) ? gnB : (N - 1);

    // ---- tile staging; fast path is wave-uniform + branch-free ----
    auto stageA = [&](int buf, int k0) {
        if (fastA && (k0 + 32 <= K)) {
            const u16* src = A + rowA * lda + k0 + sc;
            async_copy_b128(src,     &As[buf][sr][sc]);
            async_copy_b128(src + 8, &As[buf][sr][sc + 8]);
        } else {
            #pragma unroll
            for (int e = 0; e < 16; ++e) {
                const int k  = k0 + sc + e;
                const int kc = (k < K) ? k : (K - 1);
                const u16 v  = A[rowA * lda + kc];
                As[buf][sr][sc + e] = (k < K) ? v : (u16)0;
            }
        }
    };
    auto stageB = [&](int buf, int k0) {
        if (fastB && (k0 + 32 <= K)) {
            const u16* src = Bm + rowB * ldbn + k0 + sc;
            async_copy_b128(src,     &Bs[buf][sr][sc]);
            async_copy_b128(src + 8, &Bs[buf][sr][sc + 8]);
        } else {
            #pragma unroll
            for (int e = 0; e < 16; ++e) {
                const int k  = k0 + sc + e;
                const int kc = (k < K) ? k : (K - 1);
                const u16 v  = Bm[rowB * ldbn + (long long)kc * ldbk];
                Bs[buf][sr][sc + e] = (k < K) ? v : (u16)0;
            }
        }
    };

    const int nsteps = (K + 31) >> 5;

    stageA(0, 0);
    stageB(0, 0);
    wait_asynccnt0();
    __syncthreads();

    for (int s = 0; s < nsteps; ++s) {
        const int k0  = s << 5;
        const int cur = s & 1;
        const int nxt = cur ^ 1;

        // kick off next tile's async copies before touching this tile
        if (s + 1 < nsteps) {
            stageA(nxt, k0 + 32);
            stageB(nxt, k0 + 32);
        }

        // ---- build fragments per ISA 7.12.2 layouts ----
        v16bf afr[2], bfr[2];
        #pragma unroll
        for (int t = 0; t < 2; ++t) {
            // A 16x32 bf16: lane m = l15; halves hold K 0-7,16-23 / 8-15,24-31 pair-packed
            const int row = waveM * 32 + t * 16 + l15;
            const int kb  = half * 8;
            union { v16bf v; u32 d[8]; } ua;
            #pragma unroll
            for (int q = 0; q < 4; ++q) ua.d[q]     = *(const u32*)&As[cur][row][kb + 2 * q];
            #pragma unroll
            for (int q = 0; q < 4; ++q) ua.d[4 + q] = *(const u32*)&As[cur][row][kb + 16 + 2 * q];
            afr[t] = ua.v;
        }
        #pragma unroll
        for (int t = 0; t < 2; ++t) {
            // B 32x16 bf16: lane n = l15; halves hold K 0-7,16-23 / 8-15,24-31 (same as A)
            const int n  = waveN * 32 + t * 16 + l15;
            const int kb = half * 8;
            union { v16bf v; uint4 q[2]; } ub;
            ub.q[0] = *(const uint4*)&Bs[cur][n][kb];
            ub.q[1] = *(const uint4*)&Bs[cur][n][kb + 16];
            bfr[t] = ub.v;
        }

        acc[0] = __builtin_amdgcn_wmma_f32_16x16x32_bf16(false, afr[0], false, bfr[0], (short)0, acc[0], false, false);
        acc[1] = __builtin_amdgcn_wmma_f32_16x16x32_bf16(false, afr[0], false, bfr[1], (short)0, acc[1], false, false);
        acc[2] = __builtin_amdgcn_wmma_f32_16x16x32_bf16(false, afr[1], false, bfr[0], (short)0, acc[2], false, false);
        acc[3] = __builtin_amdgcn_wmma_f32_16x16x32_bf16(false, afr[1], false, bfr[1], (short)0, acc[3], false, false);

        // next tile fully in LDS + everyone done reading current tile
        wait_asynccnt0();
        __syncthreads();
    }

    // ---- epilogue: C/D layout -> lane l col n=l15, VGPR r row = r + half*8 ----
    #pragma unroll
    for (int tm = 0; tm < 2; ++tm) {
        #pragma unroll
        for (int tn = 0; tn < 2; ++tn) {
            const v8f c = acc[tm * 2 + tn];
            const int n = tileN + waveN * 32 + tn * 16 + l15;
            if (n < N) {
                const float bi = bias ? bias[n] : 0.f;
                #pragma unroll
                for (int r = 0; r < 8; ++r) {
                    const int m = tileM + waveM * 32 + tm * 16 + half * 8 + r;
                    if (m < M) {
                        float val = c[r] * alpha + bi;
                        if (resid) val += resid[(long long)m * ldr + n];
                        if (act)   val = fmaxf(val, 0.f);
                        if (outBf16) Cb[(long long)m * ldc + n] = f2bf(val);
                        else         Cf[(long long)m * ldc + n] = val;
                    }
                }
            }
        }
    }
}

// ---------------------------------------------------------------------------
// Word pooling: x[b,0,:]=hidden[b,0,:];  x[b,1+w,:]= masked mean * valid
// ---------------------------------------------------------------------------
__global__ __launch_bounds__(256)
void pool_kernel(const float* __restrict__ hidden, const int* __restrict__ widx,
                 const u8* __restrict__ gmask, const u8* __restrict__ wvalid,
                 float* __restrict__ x, u16* __restrict__ xb)
{
    const int r = blockIdx.x;          // 0 .. B*LW-1
    const int b = r / LW, p = r % LW;
    for (int c = threadIdx.x; c < HH; c += 256) {
        float val;
        if (p == 0) {
            val = hidden[(long long)b * SS * HH + c];
        } else {
            const int w = p - 1;
            const bool valid = wvalid[b * WW + w] != 0;
            float sum = 0.f, cnt = 0.f;
            #pragma unroll
            for (int g = 0; g < GG; ++g) {
                if (gmask[(b * WW + w) * GG + g]) {
                    const int t = widx[(b * WW + w) * GG + g];
                    sum += hidden[((long long)b * SS + t) * HH + c];
                    cnt += 1.f;
                }
            }
            val = valid ? sum / fmaxf(cnt, 1.f) : 0.f;
        }
        x [(long long)r * HH + c] = val;
        xb[(long long)r * HH + c] = f2bf(val);
    }
}

// ---------------------------------------------------------------------------
// Row softmax over scores (B,NH,LW,LW) with key pad mask -> bf16 probs
// ---------------------------------------------------------------------------
__global__ __launch_bounds__(128)
void softmax_rows(const float* __restrict__ sc, u16* __restrict__ pb,
                  const u8* __restrict__ wvalid)
{
    const int rid = blockIdx.x;                 // 0 .. B*NH*LW-1
    const int tid = threadIdx.x;
    const int b   = rid / (NH * LW);
    const float* row = sc + (long long)rid * LW;
    __shared__ float red[128];

    float mx = -3.4e38f;
    for (int j = tid; j < LW; j += 128) {
        const bool valid = (j == 0) || (wvalid[b * WW + j - 1] != 0);
        if (valid) mx = fmaxf(mx, row[j]);
    }
    red[tid] = mx; __syncthreads();
    for (int o = 64; o > 0; o >>= 1) { if (tid < o) red[tid] = fmaxf(red[tid], red[tid + o]); __syncthreads(); }
    mx = red[0]; __syncthreads();

    float sum = 0.f;
    for (int j = tid; j < LW; j += 128) {
        const bool valid = (j == 0) || (wvalid[b * WW + j - 1] != 0);
        sum += valid ? __expf(row[j] - mx) : 0.f;
    }
    red[tid] = sum; __syncthreads();
    for (int o = 64; o > 0; o >>= 1) { if (tid < o) red[tid] += red[tid + o]; __syncthreads(); }
    const float inv = 1.f / red[0];

    for (int j = tid; j < LW; j += 128) {
        const bool valid = (j == 0) || (wvalid[b * WW + j - 1] != 0);
        const float e = valid ? __expf(row[j] - mx) * inv : 0.f;
        pb[(long long)rid * LW + j] = f2bf(e);
    }
}

// ---------------------------------------------------------------------------
// LayerNorm over H=768, dual fp32 + bf16 output
// ---------------------------------------------------------------------------
__global__ __launch_bounds__(256)
void ln_rows(const float* __restrict__ in, const float* __restrict__ g,
             const float* __restrict__ bta, float* __restrict__ xo, u16* __restrict__ xb)
{
    const int row = blockIdx.x;
    const int tid = threadIdx.x;
    const float* p = in + (long long)row * HH;
    const float v0 = p[tid], v1 = p[tid + 256], v2 = p[tid + 512];
    __shared__ float rs[256], rq[256];
    rs[tid] = v0 + v1 + v2;
    rq[tid] = v0 * v0 + v1 * v1 + v2 * v2;
    __syncthreads();
    for (int o = 128; o > 0; o >>= 1) {
        if (tid < o) { rs[tid] += rs[tid + o]; rq[tid] += rq[tid + o]; }
        __syncthreads();
    }
    const float mean = rs[0] * (1.f / HH);
    const float var  = rq[0] * (1.f / HH) - mean * mean;
    const float inv  = rsqrtf(var + 1e-5f);
    #pragma unroll
    for (int j = 0; j < 3; ++j) {
        const int c = tid + j * 256;
        const float y = (p[c] - mean) * inv * g[c] + bta[c];
        xo[(long long)row * HH + c] = y;
        xb[(long long)row * HH + c] = f2bf(y);
    }
}

// ---------------------------------------------------------------------------
// fp32 -> bf16 bulk convert (weights)
// ---------------------------------------------------------------------------
__global__ __launch_bounds__(256)
void cvt_bf16(const float* __restrict__ in, u16* __restrict__ out, long long n)
{
    long long i = (long long)blockIdx.x * blockDim.x + threadIdx.x;
    const long long stride = (long long)gridDim.x * blockDim.x;
    for (; i < n; i += stride) out[i] = f2bf(in[i]);
}

// ---------------------------------------------------------------------------
// Classifier: logits = pooled @ cls_w^T + cls_b ; also emit pooled
// d_out layout: [B*NLAB logits][B*H pooled]
// ---------------------------------------------------------------------------
__global__ __launch_bounds__(256)
void cls_kernel(const float* __restrict__ x, const float* __restrict__ cw,
                const float* __restrict__ cb, float* __restrict__ out)
{
    const int b = blockIdx.x;
    const float* row = x + (long long)b * LW * HH;   // row 0 of sequence b
    float acc[NLAB] = {0.f, 0.f, 0.f, 0.f, 0.f, 0.f};
    for (int c = threadIdx.x; c < HH; c += 256) {
        const float v = row[c];
        out[BB * NLAB + (long long)b * HH + c] = v;  // pooled
        #pragma unroll
        for (int lab = 0; lab < NLAB; ++lab) acc[lab] += v * cw[lab * HH + c];
    }
    __shared__ float red[256];
    for (int lab = 0; lab < NLAB; ++lab) {
        red[threadIdx.x] = acc[lab]; __syncthreads();
        for (int o = 128; o > 0; o >>= 1) { if (threadIdx.x < o) red[threadIdx.x] += red[threadIdx.x + o]; __syncthreads(); }
        if (threadIdx.x == 0) out[b * NLAB + lab] = red[0] + cb[lab];
        __syncthreads();
    }
}

// ---------------------------------------------------------------------------
// Host orchestration
// ---------------------------------------------------------------------------
extern "C" void kernel_launch(void* const* d_in, const int* in_sizes, int n_in,
                              void* d_out, int out_size, void* d_ws, size_t ws_size,
                              hipStream_t stream)
{
    (void)in_sizes; (void)n_in; (void)out_size; (void)ws_size;

    const float* hidden = (const float*)d_in[0];
    const int*   widx   = (const int*)  d_in[1];
    const u8*    gmask  = (const u8*)   d_in[2];
    const u8*    wvalid = (const u8*)   d_in[3];
    const float* qkv_w  = (const float*)d_in[4];
    const float* qkv_b  = (const float*)d_in[5];
    const float* out_w  = (const float*)d_in[6];
    const float* out_b  = (const float*)d_in[7];
    const float* ff1_w  = (const float*)d_in[8];
    const float* ff1_b  = (const float*)d_in[9];
    const float* ff2_w  = (const float*)d_in[10];
    const float* ff2_b  = (const float*)d_in[11];
    const float* ln1_s  = (const float*)d_in[12];
    const float* ln1_b  = (const float*)d_in[13];
    const float* ln2_s  = (const float*)d_in[14];
    const float* ln2_b  = (const float*)d_in[15];
    const float* cls_w  = (const float*)d_in[16];
    const float* cls_b  = (const float*)d_in[17];
    float* out = (float*)d_out;

    // workspace arena
    char* ws = (char*)d_ws;
    size_t off = 0;
    auto alloc = [&](size_t bytes) -> char* {
        char* p = ws + off;
        off += (bytes + 255) & ~(size_t)255;
        return p;
    };
    float* x_f    = (float*)alloc((size_t)MR * HH * 4);
    float* tmp_f  = (float*)alloc((size_t)MR * HH * 4);
    u16*   x_b    = (u16*)  alloc((size_t)MR * HH * 2);
    u16*   qkv_bf = (u16*)  alloc((size_t)MR * 3 * HH * 2);
    float* scores = (float*)alloc((size_t)BB * NH * LW * LW * 4);
    u16*   probs  = (u16*)  alloc((size_t)BB * NH * LW * LW * 2);
    u16*   ctx_bf = (u16*)  alloc((size_t)MR * HH * 2);
    u16*   h1_bf  = (u16*)  alloc((size_t)MR * FF * 2);
    u16*   wq_bf  = (u16*)  alloc((size_t)NL * 3 * HH * HH * 2);
    u16*   wo_bf  = (u16*)  alloc((size_t)NL * HH * HH * 2);
    u16*   wf1_bf = (u16*)  alloc((size_t)NL * FF * HH * 2);
    u16*   wf2_bf = (u16*)  alloc((size_t)NL * HH * FF * 2);

    // weights -> bf16
    cvt_bf16<<<2048, 256, 0, stream>>>(qkv_w, wq_bf,  (long long)NL * 3 * HH * HH);
    cvt_bf16<<<2048, 256, 0, stream>>>(out_w, wo_bf,  (long long)NL * HH * HH);
    cvt_bf16<<<2048, 256, 0, stream>>>(ff1_w, wf1_bf, (long long)NL * FF * HH);
    cvt_bf16<<<2048, 256, 0, stream>>>(ff2_w, wf2_bf, (long long)NL * HH * FF);

    // build x (fp32 + bf16)
    pool_kernel<<<MR, 256, 0, stream>>>(hidden, widx, gmask, wvalid, x_f, x_b);

    auto gemm = [&](const u16* A, long long lda, long long sAo, long long sAi,
                    const u16* Bm, long long ldbn, long long ldbk, long long sBo, long long sBi,
                    const float* bias, const float* resid, long long ldr,
                    void* C, long long ldc, long long sCo, long long sCi,
                    int M_, int N_, int K_, float alpha, int act, int obf,
                    int batchInner, int nz) {
        dim3 grd((N_ + 63) / 64, (M_ + 63) / 64, nz);
        gemm_bf16_wmma<<<grd, 128, 0, stream>>>(A, lda, sAo, sAi, Bm, ldbn, ldbk, sBo, sBi,
                                                bias, resid, ldr, C, ldc, sCo, sCi,
                                                M_, N_, K_, alpha, act, obf, batchInner);
    };

    const float scale = 0.07216878364870323f;   // 1/sqrt(192)

    for (int l = 0; l < NL; ++l) {
        // QKV projection -> bf16 qkv (M x 2304)
        gemm(x_b, HH, 0, 0,
             wq_bf + (size_t)l * 3 * HH * HH, HH, 1, 0, 0,
             qkv_b + l * 3 * HH, nullptr, 0,
             qkv_bf, 3 * HH, 0, 0,
             MR, 3 * HH, HH, 1.f, 0, /*bf16 out*/1, 1, 1);

        // scores = scale * q @ k^T  (batched over B*NH)
        gemm(qkv_bf, 3 * HH, (long long)LW * 3 * HH, HD,
             qkv_bf + HH, 3 * HH, 1, (long long)LW * 3 * HH, HD,
             nullptr, nullptr, 0,
             scores, LW, (long long)NH * LW * LW, (long long)LW * LW,
             LW, LW, HD, scale, 0, /*f32 out*/0, NH, BB * NH);

        softmax_rows<<<BB * NH * LW, 128, 0, stream>>>(scores, probs, wvalid);

        // ctx = probs @ v   (v accessed transposed via ldbk = 3H)
        gemm(probs, LW, (long long)NH * LW * LW, (long long)LW * LW,
             qkv_bf + 2 * HH, 1, 3 * HH, (long long)LW * 3 * HH, HD,
             nullptr, nullptr, 0,
             ctx_bf, HH, (long long)LW * HH, HD,
             LW, HD, LW, 1.f, 0, /*bf16 out*/1, NH, BB * NH);

        // attn out proj + residual -> tmp (fp32), then LN1 -> x
        gemm(ctx_bf, HH, 0, 0,
             wo_bf + (size_t)l * HH * HH, HH, 1, 0, 0,
             out_b + l * HH, x_f, HH,
             tmp_f, HH, 0, 0,
             MR, HH, HH, 1.f, 0, 0, 1, 1);
        ln_rows<<<MR, 256, 0, stream>>>(tmp_f, ln1_s + l * HH, ln1_b + l * HH, x_f, x_b);

        // FF1 (+ReLU) -> bf16 h1
        gemm(x_b, HH, 0, 0,
             wf1_bf + (size_t)l * FF * HH, HH, 1, 0, 0,
             ff1_b + l * FF, nullptr, 0,
             h1_bf, FF, 0, 0,
             MR, FF, HH, 1.f, /*relu*/1, 1, 1, 1);

        // FF2 + residual -> tmp, then LN2 -> x
        gemm(h1_bf, FF, 0, 0,
             wf2_bf + (size_t)l * HH * FF, FF, 1, 0, 0,
             ff2_b + l * HH, x_f, HH,
             tmp_f, HH, 0, 0,
             MR, HH, FF, 1.f, 0, 0, 1, 1);
        ln_rows<<<MR, 256, 0, stream>>>(tmp_f, ln2_s + l * HH, ln2_b + l * HH, x_f, x_b);
    }

    // logits + pooled
    cls_kernel<<<BB, 256, 0, stream>>>(x_f, cls_w, cls_b, out);
}
